// SymmetricSelfAttention_14680198218252
// MI455X (gfx1250) — matrix operand order repaired and checked by reading, not compiled, not run on hardware
//
#include <hip/hip_runtime.h>
#include <hip/hip_bf16.h>

// ---------------------------------------------------------------------------
// Symmetric self-attention, fused flash-style, bf16 WMMA (gfx1250 / MI455X)
//
// B=2, N=2048, HIDDEN=256, HEADS=8, HEAD_DIM=32, SCALE=1/sqrt(32)
//
// Design notes (MI455X):
//  * Never materialize [B,H,N,N]: all HBM traffic ~20MB (L2-resident, 192MB).
//  * bias MLP factored: relu((c_i.w1+b1) - (c_j.w1)) -> per-token 16-vectors.
//  * pos_b2/sym_b2 are constant along softmax axis -> cancel, skipped.
//  * Per-pair hidden relu computed ONCE for all 8 heads (LDS shared).
//  * HEAD_DIM=32 == one v_wmma_f32_16x16x32_bf16 K step.
//  * Weights pre-packed once into WMMA B-operand bf16 layout (2x b128/lane).
//  * V written by the QKV kernel directly in B-operand layout (vpack), so the
//    PV stage needs only contiguous 16-byte loads in the 64-iteration loop.
//  * f32->bf16 via native cast (v_cvt_*bf16*), not bit-twiddling.
//  * global_prefetch_b8 for next j-step K/V while bias MLP runs.
// ---------------------------------------------------------------------------

#define NTOK   2048
#define BATCH  2
#define HID    256
#define QKVC   768
#define NHEAD  8
#define HDIM   32
#define NJBLK  (NTOK / 32)
#define ATT_SCALE 0.17677669529663689f  // 32^-0.5

typedef __attribute__((ext_vector_type(16))) __bf16 v16bf;
typedef __attribute__((ext_vector_type(8)))  float  v8f;
typedef __attribute__((ext_vector_type(4)))  unsigned int v4u;

union BF16x16 {
  v16bf v;
  v4u   q2[2];
  __bf16 e[16];
};

__device__ inline __bf16 f2bf(float f) { return (__bf16)f; }

__device__ inline v8f wmma_bf16(v16bf a, v16bf b, v8f c) {
  return __builtin_amdgcn_wmma_f32_16x16x32_bf16(
      /*neg_a=*/false, a, /*neg_b=*/false, b,
      /*c_mod=*/(short)0, c, /*reuse_a=*/false, /*reuse_b=*/false);
}

// ---------------------------------------------------------------------------
// Kernel 0: pack a [256, ncols] f32 weight matrix into WMMA B-operand bf16
// layout: out[ct][k][lane][t] = W[k*32 + (lane>>4)*16 + t][ct*16 + (lane&15)]
// One-time cost; afterwards every B operand is two contiguous b128 loads.
// ---------------------------------------------------------------------------
__global__ __launch_bounds__(256) void pack_w_kernel(
    const float* __restrict__ W, __bf16* __restrict__ out, int ncolTiles, int ncols)
{
  int o = blockIdx.x * 256 + threadIdx.x;
  int total = ncolTiles * 8 * 32 * 16;
  if (o >= total) return;
  int t    = o & 15;
  int lane = (o >> 4) & 31;
  int k    = (o >> 9) & 7;
  int ct   = o >> 12;
  int row  = k * 32 + (lane >> 4) * 16 + t;
  int col  = ct * 16 + (lane & 15);
  out[o] = f2bf(W[(size_t)row * ncols + col]);
}

// ---------------------------------------------------------------------------
// Kernel 1: qkv = x @ W_qkv ([4096,256]x[256,768]), W pre-packed bf16.
// Q/K columns -> row-major qkv buffer; V columns -> vpack B-operand layout.
// grid(256), block(256)=8 waves; each wave owns 6 of 48 column tiles.
// ---------------------------------------------------------------------------
__global__ __launch_bounds__(256) void qkv_proj_kernel(
    const float* __restrict__ x, const __bf16* __restrict__ Wp,
    __bf16* __restrict__ qkv, __bf16* __restrict__ vpack)
{
  __shared__ __bf16 ldsA[16][HID];                 // 8 KB, x tile as bf16
  const int m0  = blockIdx.x * 16;
  const int tid = threadIdx.x;
  for (int e = tid; e < 16 * HID; e += 256) {
    int r = e >> 8, c = e & 255;
    ldsA[r][c] = f2bf(x[(size_t)(m0 + r) * HID + c]);
  }
  __syncthreads();
  const int wave = tid >> 5, lane = tid & 31;
  const int ln = lane & 15, hi = lane >> 4;
  for (int t = 0; t < 6; ++t) {
    const int ct = wave * 6 + t;
    const int c0 = ct * 16;
    v8f acc = {};
    for (int k = 0; k < 8; ++k) {
      BF16x16 a;                                   // A: row=ln, K split by half
      const __bf16* ap = &ldsA[ln][k * 32 + hi * 16];
      a.q2[0] = *(const v4u*)(ap);
      a.q2[1] = *(const v4u*)(ap + 8);
      BF16x16 b;                                   // B: packed, contiguous
      const __bf16* wp = Wp + ((size_t)(ct * 8 + k) * 32 + lane) * 16;
      b.q2[0] = *(const v4u*)(wp);
      b.q2[1] = *(const v4u*)(wp + 8);
      acc = wmma_bf16(a.v, b.v, acc);
    }
    if (c0 < 2 * HID) {
      // Q or K: row-major store. C: row=r+hi*8, col=ln
      #pragma unroll
      for (int r = 0; r < 8; ++r)
        qkv[(size_t)(m0 + r + hi * 8) * QKVC + c0 + ln] = f2bf(acc[r]);
    } else {
      // V: store directly in WMMA B-operand layout for the PV stage.
      const int h  = (c0 - 2 * HID) >> 5;
      const int dh = ((c0 - 2 * HID) >> 4) & 1;
      #pragma unroll
      for (int r = 0; r < 8; ++r) {
        int rowg = m0 + r + hi * 8;                // flat b*N + j
        int b_   = rowg >> 11;
        int j    = rowg & (NTOK - 1);
        int jblk = j >> 5, jj = j & 31;
        int laneB = (jj >> 4) * 16 + ln;           // B-operand lane
        size_t dst = ((((size_t)(b_ * NHEAD + h) * NJBLK + jblk) * 2 + dh) * 32
                      + laneB) * 16 + (jj & 15);
        vpack[dst] = f2bf(acc[r]);
      }
    }
  }
}

// ---------------------------------------------------------------------------
// Kernel 2: per-token 16-vectors for the factored pair-bias MLPs.
//  paD[i] = c_i.pos_w1 + pos_b1      pbD[j] = c_j.pos_w1
//  paS[i] = c_i.sym_w1 + sym_b1      pbS[j] = (reflect*c_j).sym_w1
// ---------------------------------------------------------------------------
__global__ __launch_bounds__(256) void coords_proj_kernel(
    const float* __restrict__ coords,
    const float* __restrict__ pw1, const float* __restrict__ pb1,
    const float* __restrict__ sw1, const float* __restrict__ sb1,
    float* __restrict__ paD, float* __restrict__ pbD,
    float* __restrict__ paS, float* __restrict__ pbS)
{
  int idx = blockIdx.x * 256 + threadIdx.x;        // b*N+i, 0..4095
  if (idx >= BATCH * NTOK) return;
  float c0 = coords[idx * 3 + 0];
  float c1 = coords[idx * 3 + 1];
  float c2 = coords[idx * 3 + 2];
  #pragma unroll
  for (int k = 0; k < 16; ++k) {
    float d  = c0 * pw1[k] + c1 * pw1[16 + k] + c2 * pw1[32 + k];
    float s  = c0 * sw1[k] + c1 * sw1[16 + k] + c2 * sw1[32 + k];
    float sr = -c0 * sw1[k] + c1 * sw1[16 + k] + c2 * sw1[32 + k];
    paD[idx * 16 + k] = d + pb1[k];
    pbD[idx * 16 + k] = d;
    paS[idx * 16 + k] = s + sb1[k];
    pbS[idx * 16 + k] = sr;                        // reflect = [-1,1,1]
  }
}

// ---------------------------------------------------------------------------
// Kernel 3: fused flash attention with shared pair-bias.
// grid(256) = (B, i-tile of 16 rows); block(256) = 8 waves = 8 heads.
// Per 32-wide j step: bias tile computed once for all heads (LDS), then each
// wave: 2 WMMA (S), online softmax, 2 WMMA (PV).
// ---------------------------------------------------------------------------
__global__ __launch_bounds__(256) void attn_kernel(
    const __bf16* __restrict__ qkv, const __bf16* __restrict__ vpack,
    const float* __restrict__ paD, const float* __restrict__ pbD,
    const float* __restrict__ paS, const float* __restrict__ pbS,
    const float* __restrict__ pw2, const float* __restrict__ sw2,
    __bf16* __restrict__ attn_out)
{
  __shared__ float  sBias[NHEAD][16][32];          // 16 KB
  __shared__ float  sPaD[16][16], sPaS[16][16];    //  2 KB
  __shared__ float  sPbD[32][16], sPbS[32][16];    //  4 KB
  __shared__ float  sW2D[16][NHEAD], sW2S[16][NHEAD];
  __shared__ __bf16 sP[NHEAD][16][32];             //  8 KB (P relayout C->A)

  const int b   = blockIdx.x >> 7;                 // 128 i-tiles per batch
  const int i0  = (blockIdx.x & 127) * 16;
  const int tid = threadIdx.x;
  const int h   = tid >> 5;                        // wave == head
  const int lane = tid & 31;
  const int ln = lane & 15, hi = lane >> 4;

  if (tid < 128) {
    sW2D[tid >> 3][tid & 7] = pw2[tid];
    sW2S[tid >> 3][tid & 7] = sw2[tid];
  }
  for (int e = tid; e < 16 * 16; e += 256) {
    int r = e >> 4, k = e & 15;
    size_t base = ((size_t)(b * NTOK + i0 + r)) * 16 + k;
    sPaD[r][k] = paD[base];
    sPaS[r][k] = paS[base];
  }

  // Q as WMMA A operand (16x32, K=HDIM), head h, rows i0..i0+15
  BF16x16 qa;
  {
    const __bf16* qp = qkv + (size_t)(b * NTOK + i0 + ln) * QKVC + h * HDIM + hi * 16;
    qa.q2[0] = *(const v4u*)(qp);
    qa.q2[1] = *(const v4u*)(qp + 8);
  }

  v8f O0 = {}, O1 = {};
  float mrow[8], lrow[8];
  #pragma unroll
  for (int r = 0; r < 8; ++r) { mrow[r] = -1e30f; lrow[r] = 0.0f; }

  __syncthreads();

  for (int j0 = 0; j0 < NTOK; j0 += 32) {
    // ---- prefetch next j block's K rows and V pack block (CDNA5 path) ----
    if (j0 + 32 < NTOK) {
      __builtin_prefetch(qkv + (size_t)(b * NTOK + j0 + 32 + ln) * QKVC
                             + HID + h * HDIM, 0, 0);
      __builtin_prefetch(vpack + ((((size_t)(b * NHEAD + h) * NJBLK
                             + ((j0 >> 5) + 1)) * 2) * 32 + lane) * 16, 0, 0);
    }

    // ---- cooperative: stage pb tiles for this j block ----
    for (int e = tid; e < 32 * 16; e += 256) {
      int r = e >> 4, k = e & 15;
      size_t base = ((size_t)(b * NTOK + j0 + r)) * 16 + k;
      sPbD[r][k] = pbD[base];
      sPbS[r][k] = pbS[base];
    }
    __syncthreads();

    // ---- cooperative: pair-bias MLP, one pass for ALL 8 heads ----
    // (pos_b2 / sym_b2 omitted: constant along softmax axis -> cancel)
    for (int p = tid; p < 16 * 32; p += 256) {
      int pi = p >> 5, pj = p & 31;
      float acc[NHEAD];
      #pragma unroll
      for (int hh = 0; hh < NHEAD; ++hh) acc[hh] = 0.0f;
      #pragma unroll
      for (int k = 0; k < 16; ++k) {
        float hd = sPaD[pi][k] - sPbD[pj][k]; hd = hd > 0.0f ? hd : 0.0f;
        float hs = sPaS[pi][k] - sPbS[pj][k]; hs = hs > 0.0f ? hs : 0.0f;
        #pragma unroll
        for (int hh = 0; hh < NHEAD; ++hh)
          acc[hh] += hd * sW2D[k][hh] + hs * sW2S[k][hh];
      }
      #pragma unroll
      for (int hh = 0; hh < NHEAD; ++hh) sBias[hh][pi][pj] = acc[hh];
    }
    __syncthreads();

    // ---- per-wave: S = Q K^T (two 16x16 tiles, one WMMA each) ----
    BF16x16 kb0, kb1;                              // B: col=jj(=ln), K=d
    {
      const __bf16* kp = qkv + (size_t)(b * NTOK + j0 + ln) * QKVC
                             + HID + h * HDIM + hi * 16;
      kb0.q2[0] = *(const v4u*)(kp);
      kb0.q2[1] = *(const v4u*)(kp + 8);
      const __bf16* kp1 = kp + (size_t)16 * QKVC;
      kb1.q2[0] = *(const v4u*)(kp1);
      kb1.q2[1] = *(const v4u*)(kp1 + 8);
    }
    v8f z = {};
    v8f S0 = wmma_bf16(qa.v, kb0.v, z);
    v8f S1 = wmma_bf16(qa.v, kb1.v, z);

    // ---- online softmax (rows live in 16-lane halves) ----
    float corr[8];
    v8f P0, P1;
    #pragma unroll
    for (int r = 0; r < 8; ++r) {
      int row = r + hi * 8;
      float s0 = S0[r] * ATT_SCALE + sBias[h][row][ln];
      float s1 = S1[r] * ATT_SCALE + sBias[h][row][ln + 16];
      float mx = fmaxf(s0, s1);
      #pragma unroll
      for (int off = 8; off >= 1; off >>= 1)
        mx = fmaxf(mx, __shfl_xor(mx, off, 32));
      float mnew = fmaxf(mrow[r], mx);
      float p0 = __expf(s0 - mnew);
      float p1 = __expf(s1 - mnew);
      float ps = p0 + p1;
      #pragma unroll
      for (int off = 8; off >= 1; off >>= 1)
        ps += __shfl_xor(ps, off, 32);
      float c = __expf(mrow[r] - mnew);
      corr[r] = c;
      lrow[r] = lrow[r] * c + ps;
      mrow[r] = mnew;
      P0[r] = p0;
      P1[r] = p1;
    }
    #pragma unroll
    for (int r = 0; r < 8; ++r) { O0[r] *= corr[r]; O1[r] *= corr[r]; }

    // ---- P: C layout -> A layout via wave-private LDS ----
    #pragma unroll
    for (int r = 0; r < 8; ++r) {
      int row = r + hi * 8;
      sP[h][row][ln]      = f2bf(P0[r]);
      sP[h][row][ln + 16] = f2bf(P1[r]);
    }
    BF16x16 pA;
    {
      const __bf16* pp = &sP[h][ln][hi * 16];
      pA.q2[0] = *(const v4u*)(pp);
      pA.q2[1] = *(const v4u*)(pp + 8);
    }

    // ---- V from vpack (B-operand layout, contiguous b128 loads) ----
    BF16x16 vb0, vb1;
    {
      const __bf16* vp = vpack + ((((size_t)(b * NHEAD + h) * NJBLK
                            + (j0 >> 5)) * 2) * 32 + lane) * 16;
      vb0.q2[0] = *(const v4u*)(vp);
      vb0.q2[1] = *(const v4u*)(vp + 8);
      const __bf16* vp1 = vp + 32 * 16;            // d-half 1
      vb1.q2[0] = *(const v4u*)(vp1);
      vb1.q2[1] = *(const v4u*)(vp1 + 8);
    }
    O0 = wmma_bf16(pA.v, vb0.v, O0);
    O1 = wmma_bf16(pA.v, vb1.v, O1);

    __syncthreads();                               // protect sPb/sBias reuse
  }

  // ---- finalize: O /= l, write bf16 [B,N,H*D] ----
  #pragma unroll
  for (int r = 0; r < 8; ++r) {
    int row = r + hi * 8;
    float inv = 1.0f / lrow[r];
    size_t o = (size_t)(b * NTOK + i0 + row) * HID + h * HDIM;
    attn_out[o + ln]      = f2bf(O0[r] * inv);
    attn_out[o + 16 + ln] = f2bf(O1[r] * inv);
  }
}

// ---------------------------------------------------------------------------
// Kernel 4: out = attn @ W_out + b_out ([4096,256]x[256,256] -> f32), W packed.
// grid(256,2), block(256): wave handles one of 16 column tiles.
// ---------------------------------------------------------------------------
__global__ __launch_bounds__(256) void out_proj_kernel(
    const __bf16* __restrict__ a, const __bf16* __restrict__ Wp,
    const float* __restrict__ bias, float* __restrict__ out)
{
  const int m0 = blockIdx.x * 16;
  const int tid = threadIdx.x;
  const int wave = tid >> 5, lane = tid & 31;
  const int ln = lane & 15, hi = lane >> 4;
  const int ct = blockIdx.y * 8 + wave;
  const int c0 = ct * 16;
  v8f acc = {};
  for (int k = 0; k < 8; ++k) {
    BF16x16 av;
    const __bf16* ap = a + (size_t)(m0 + ln) * HID + k * 32 + hi * 16;
    av.q2[0] = *(const v4u*)(ap);
    av.q2[1] = *(const v4u*)(ap + 8);
    BF16x16 bv;
    const __bf16* wp = Wp + ((size_t)(ct * 8 + k) * 32 + lane) * 16;
    bv.q2[0] = *(const v4u*)(wp);
    bv.q2[1] = *(const v4u*)(wp + 8);
    acc = wmma_bf16(av.v, bv.v, acc);
  }
  float bo = bias[c0 + ln];
  #pragma unroll
  for (int r = 0; r < 8; ++r)
    out[(size_t)(m0 + r + hi * 8) * HID + c0 + ln] = acc[r] + bo;
}

// ---------------------------------------------------------------------------
extern "C" void kernel_launch(void* const* d_in, const int* in_sizes, int n_in,
                              void* d_out, int out_size, void* d_ws, size_t ws_size,
                              hipStream_t stream) {
  (void)in_sizes; (void)n_in; (void)out_size; (void)ws_size;
  const float* x      = (const float*)d_in[0];
  const float* coords = (const float*)d_in[1];
  const float* W_qkv  = (const float*)d_in[2];
  const float* W_out  = (const float*)d_in[3];
  const float* b_out  = (const float*)d_in[4];
  const float* pos_w1 = (const float*)d_in[5];
  const float* pos_b1 = (const float*)d_in[6];
  const float* pos_w2 = (const float*)d_in[7];
  // d_in[8] = pos_b2: cancels in softmax, unused
  const float* sym_w1 = (const float*)d_in[9];
  const float* sym_b1 = (const float*)d_in[10];
  const float* sym_w2 = (const float*)d_in[11];
  // d_in[12] = sym_b2: cancels in softmax, unused

  // workspace layout (~12 MB)
  char* ws = (char*)d_ws;
  __bf16* qkv    = (__bf16*)ws;                               // 4096*768
  __bf16* attn   = qkv + (size_t)BATCH * NTOK * QKVC;         // 4096*256
  __bf16* vpack  = attn + (size_t)BATCH * NTOK * HID;         // 2*8*64*2*32*16
  __bf16* WpQKV  = vpack + (size_t)BATCH * NHEAD * NJBLK * 2 * 32 * 16;
  __bf16* WpOut  = WpQKV + (size_t)48 * 8 * 32 * 16;          // 196608
  float*  paD    = (float*)(WpOut + (size_t)16 * 8 * 32 * 16);
  float*  pbD    = paD + (size_t)BATCH * NTOK * 16;
  float*  paS    = pbD + (size_t)BATCH * NTOK * 16;
  float*  pbS    = paS + (size_t)BATCH * NTOK * 16;

  pack_w_kernel<<<dim3(768), 256, 0, stream>>>(W_qkv, WpQKV, 48, QKVC);
  pack_w_kernel<<<dim3(256), 256, 0, stream>>>(W_out, WpOut, 16, HID);
  qkv_proj_kernel<<<dim3(BATCH * NTOK / 16), 256, 0, stream>>>(x, WpQKV, qkv, vpack);
  coords_proj_kernel<<<dim3(BATCH * NTOK / 256), 256, 0, stream>>>(
      coords, pos_w1, pos_b1, sym_w1, sym_b1, paD, pbD, paS, pbS);
  attn_kernel<<<dim3(BATCH * (NTOK / 16)), 256, 0, stream>>>(
      qkv, vpack, paD, pbD, paS, pbS, pos_w2, sym_w2, attn);
  out_proj_kernel<<<dim3(BATCH * NTOK / 16, 2), 256, 0, stream>>>(
      attn, WpOut, b_out, (float*)d_out);
}
